// FGCL4Rec_27693949125370
// MI455X (gfx1250) — matrix operand order repaired and checked by reading, not compiled, not run on hardware
//
#include <hip/hip_runtime.h>

#define NITEMS 6001
#define NPAD   6016
#define DDIM   64
#define BB     32
#define LL     200
#define NPOS   (BB*LL)      // 6400
#define KCHUNKS (NPAD/32)   // 188

typedef __attribute__((ext_vector_type(16))) __bf16 v16bf;
typedef __attribute__((ext_vector_type(8)))  float  v8f;

union BFVec { v16bf v; unsigned u[8]; unsigned short s[16]; };

__device__ __forceinline__ unsigned short f2bf(float f) {
  union { float f; unsigned u; } x; x.f = f;
  unsigned r = x.u + 0x7FFFu + ((x.u >> 16) & 1u);   // round-to-nearest-even
  return (unsigned short)(r >> 16);
}
__device__ __forceinline__ float lrelu(float x){ return x > 0.f ? x : 0.01f * x; }
__device__ __forceinline__ float sigm(float x){ return 1.f / (1.f + __expf(-x)); }

// ---------------------------------------------------------------------------
// K1: h = item_emb @ W_item ; emit hT (bf16, [64][NPAD]) and wh1/wh2 ([NPAD])
// ---------------------------------------------------------------------------
__global__ __launch_bounds__(256) void k1_hproj(
    const float* __restrict__ item_emb, const float* __restrict__ W,
    const float* __restrict__ a_item, unsigned short* __restrict__ hT,
    float* __restrict__ wh1, float* __restrict__ wh2)
{
  __shared__ float sw[64*64];
  __shared__ float r1[256], r2[256];
  for (int i = threadIdx.x; i < 4096; i += 256) sw[i] = W[i];
  __syncthreads();
  const int tid = threadIdx.x;
  const int pos = blockIdx.x * 4 + (tid >> 6);
  const int d   = tid & 63;
  float acc = 0.f;
  if (pos < NITEMS) {
    const float* e = item_emb + (size_t)pos * 64;
    #pragma unroll
    for (int k = 0; k < 64; ++k) acc = fmaf(e[k], sw[k*64 + d], acc);
  }
  hT[(size_t)d * NPAD + pos] = f2bf(acc);
  r1[tid] = acc * a_item[d];
  r2[tid] = acc * a_item[64 + d];
  __syncthreads();
  for (int s = 32; s > 0; s >>= 1) {
    if (d < s) { r1[tid] += r1[tid + s]; r2[tid] += r2[tid + s]; }
    __syncthreads();
  }
  if (d == 0) { wh1[pos] = r1[tid]; wh2[pos] = r2[tid]; }
}

// ---------------------------------------------------------------------------
// K2: fused masked-softmax GAT:  gat = softmax(mask(lrelu(wh1+wh2^T))) @ h
//                                trans = adj @ h        (bf16 WMMA, f32 acc)
// One block = 16 output rows; 8 waves split K; stats recomputed from rank-1 e.
// Hot path (375/376 blocks x 187/188 chunks) is branchless.
// ---------------------------------------------------------------------------
__global__ __launch_bounds__(256) void k2_gat(
    const float* __restrict__ adj, const unsigned short* __restrict__ hT,
    const float* __restrict__ wh1, const float* __restrict__ wh2,
    float* __restrict__ trans, float* __restrict__ gat)
{
  __shared__ float s_m[16], s_inv[16], s_w1[16];
  __shared__ float red[256];
  __shared__ unsigned short tA[8][16][34];   // per-wave attn bf16 tile
  __shared__ unsigned short tT[8][16][34];   // per-wave adj  bf16 tile
  __shared__ float accG[16][64];
  __shared__ float accT[16][64];

  const int tid = threadIdx.x;
  const int i0  = blockIdx.x * 16;

  // ---------- phase 1: per-row softmax stats (max, 1/sum) ----------
  const int r   = tid >> 4;     // 0..15 row-in-tile
  const int l16 = tid & 15;
  const int row = i0 + r;
  const float w1 = wh1[row];
  float mx = -9.0e15f;
  if (row < NITEMS) {
    const float* ar = adj + (size_t)row * NITEMS;
    for (int j = l16; j < NITEMS; j += 16) {
      __builtin_prefetch(ar + j + 1024, 0, 1);
      float a = ar[j];
      float e = (a > 0.f) ? lrelu(w1 + wh2[j]) : -9.0e15f;
      mx = fmaxf(mx, e);
    }
  }
  red[tid] = mx;
  __syncthreads();
  for (int s = 8; s > 0; s >>= 1) {
    if (l16 < s) red[tid] = fmaxf(red[tid], red[tid + s]);
    __syncthreads();
  }
  const float m = red[r << 4];
  float se = 0.f;
  if (row < NITEMS) {
    const float* ar = adj + (size_t)row * NITEMS;
    for (int j = l16; j < NITEMS; j += 16) {
      float a = ar[j];
      float e = (a > 0.f) ? lrelu(w1 + wh2[j]) : -9.0e15f;
      se += __expf(e - m);
    }
  }
  __syncthreads();
  red[tid] = se;
  __syncthreads();
  for (int s = 8; s > 0; s >>= 1) {
    if (l16 < s) red[tid] += red[tid + s];
    __syncthreads();
  }
  if (l16 == 0) {
    s_m[r]   = m;
    float dn = red[tid];
    s_inv[r] = dn > 0.f ? 1.f / dn : 0.f;
    s_w1[r]  = w1;
  }
  for (int i = tid; i < 16*64; i += 256) { (&accG[0][0])[i] = 0.f; (&accT[0][0])[i] = 0.f; }
  __syncthreads();

  // ---------- phase 2: bf16 WMMA accumulation over K ----------
  const int wave = tid >> 5;
  const int lane = tid & 31;
  const int Mrow = lane & 15;
  const int half = lane >> 4;
  const bool fullBlock = (i0 + 16 <= NITEMS);           // uniform
  const size_t rowBase = (size_t)i0 * NITEMS;

  v8f cg[4], ct[4];
  #pragma unroll
  for (int n = 0; n < 4; ++n) {
    #pragma unroll
    for (int q = 0; q < 8; ++q) { cg[n][q] = 0.f; ct[n][q] = 0.f; }
  }

  for (int kc = wave; kc < KCHUNKS; kc += 8) {
    const int k0 = kc * 32;
    const int j  = k0 + lane;

    if (fullBlock && (k0 + 32 <= NITEMS)) {
      // ---- branchless hot path: all rows & columns valid ----
      const float  w2j = wh2[j];
      const float* ap  = adj + rowBase + j;
      #pragma unroll
      for (int rr = 0; rr < 16; ++rr) {
        const float a    = ap[(size_t)rr * NITEMS];     // imm-offset, clauseable
        const float e    = (a > 0.f) ? lrelu(s_w1[rr] + w2j) : -9.0e15f;
        const float attn = __expf(e - s_m[rr]) * s_inv[rr];
        tA[wave][rr][lane] = f2bf(attn);
        tT[wave][rr][lane] = f2bf(a);
      }
    } else {
      // ---- edge path: clamp addresses, cndmask values (no exec branching) ----
      const bool  jv  = (j < NITEMS);
      const int   jc  = jv ? j : (NITEMS - 1);
      const float w2j = jv ? wh2[j] : 0.f;
      #pragma unroll
      for (int rr = 0; rr < 16; ++rr) {
        const int rowr = i0 + rr;
        const int rc   = (rowr < NITEMS) ? rowr : (NITEMS - 1);
        float a = adj[(size_t)rc * NITEMS + jc];
        if (!(jv && rowr < NITEMS)) a = 0.f;
        const float e    = (a > 0.f) ? lrelu(s_w1[rr] + w2j) : -9.0e15f;
        const float attn = jv ? __expf(e - s_m[rr]) * s_inv[rr] : 0.f;
        tA[wave][rr][lane] = f2bf(attn);
        tT[wave][rr][lane] = f2bf(a);
      }
    }

    // A fragments: 16x32 bf16 layout (lane=M; VGPR v -> K pair)
    BFVec Aa, At;
    #pragma unroll
    for (int v = 0; v < 8; ++v) {
      const int K = 2*v + (half ? 16 : 0);
      Aa.u[v] = (unsigned)tA[wave][Mrow][K] | ((unsigned)tA[wave][Mrow][K+1] << 16);
      At.u[v] = (unsigned)tT[wave][Mrow][K] | ((unsigned)tT[wave][Mrow][K+1] << 16);
    }
    // B fragments from bf16 hT (pairs {K,K+1} contiguous) + 8 WMMAs
    #pragma unroll
    for (int n = 0; n < 4; ++n) {
      const int Ncol = n * 16 + Mrow;
      const unsigned* hp = (const unsigned*)(hT + (size_t)Ncol * NPAD + k0);
      BFVec Bf;
      #pragma unroll
      for (int v = 0; v < 8; ++v) Bf.u[v] = hp[v + (half ? 8 : 0)];
      cg[n] = __builtin_amdgcn_wmma_f32_16x16x32_bf16(false, Aa.v, false, Bf.v,
                                                      (short)0, cg[n], false, false);
      ct[n] = __builtin_amdgcn_wmma_f32_16x16x32_bf16(false, At.v, false, Bf.v,
                                                      (short)0, ct[n], false, false);
    }
  }

  // cross-wave reduction of C tiles (lane -> row = vgpr + 8*half, col = 16n+M)
  #pragma unroll
  for (int rr = 0; rr < 8; ++rr) {
    const int orow = rr + half * 8;
    #pragma unroll
    for (int n = 0; n < 4; ++n) {
      atomicAdd(&accG[orow][n*16 + Mrow], cg[n][rr]);
      atomicAdd(&accT[orow][n*16 + Mrow], ct[n][rr]);
    }
  }
  __syncthreads();
  for (int i = tid; i < 16*64; i += 256) {
    const int rr = i >> 6, c = i & 63;
    gat  [(size_t)(i0 + rr) * 64 + c] = accG[rr][c];
    trans[(size_t)(i0 + rr) * 64 + c] = accT[rr][c];
  }
}

// ---------------------------------------------------------------------------
// K3: gathers + coff gate + positional add
// ---------------------------------------------------------------------------
__global__ __launch_bounds__(256) void k3_seqs(
    const int* __restrict__ logq, const float* __restrict__ item_emb,
    const float* __restrict__ pos_emb, const float* __restrict__ coc,
    const float* __restrict__ con, const float* __restrict__ gat,
    const float* __restrict__ trans, float* __restrict__ seqs,
    float* __restrict__ seqsp)
{
  __shared__ float sc[4096], sn[4096];
  for (int i = threadIdx.x; i < 4096; i += 256) { sc[i] = coc[i]; sn[i] = con[i]; }
  __syncthreads();
  const int p = blockIdx.x * 4 + (threadIdx.x >> 6);
  const int d = threadIdx.x & 63;
  const int idx = logq[p];
  const float* g = gat   + (size_t)idx * 64;
  const float* t = trans + (size_t)idx * 64;
  float a = 0.f;
  #pragma unroll
  for (int k = 0; k < 64; ++k) { a = fmaf(g[k], sc[k*64 + d], a); a = fmaf(t[k], sn[k*64 + d], a); }
  const float c  = sigm(a);
  const float sv = c * g[d] + (1.f - c) * t[d] + item_emb[(size_t)idx * 64 + d];
  seqs[(size_t)p * 64 + d] = sv;
  const int l = p % LL;
  seqsp[(size_t)p * 64 + d] = sv + ((idx != 0) ? pos_emb[l*64 + d] : 0.f);
}

// ---------------------------------------------------------------------------
// K4: M1 = seqs_pos @ W_1 ; M2 = seqs_pos @ W_2
// ---------------------------------------------------------------------------
__global__ __launch_bounds__(256) void k4_m12(
    const float* __restrict__ sp, const float* __restrict__ W1,
    const float* __restrict__ W2, float* __restrict__ M1, float* __restrict__ M2)
{
  __shared__ float sw1[4096], sw2[4096];
  for (int i = threadIdx.x; i < 4096; i += 256) { sw1[i] = W1[i]; sw2[i] = W2[i]; }
  __syncthreads();
  const int p = blockIdx.x * 4 + (threadIdx.x >> 6);
  const int d = threadIdx.x & 63;
  const float* x = sp + (size_t)p * 64;
  float a1 = 0.f, a2 = 0.f;
  #pragma unroll
  for (int k = 0; k < 64; ++k) { const float xv = x[k]; a1 = fmaf(xv, sw1[k*64+d], a1); a2 = fmaf(xv, sw2[k*64+d], a2); }
  M1[(size_t)p*64 + d] = a1;
  M2[(size_t)p*64 + d] = a2;
}

// ---------------------------------------------------------------------------
// K5: causal sigmoid score + final = score @ seqs (one block per (b,q))
// ---------------------------------------------------------------------------
__global__ __launch_bounds__(64) void k5_attn(
    const float* __restrict__ M1, const float* __restrict__ M2,
    const float* __restrict__ seqs, const float* __restrict__ b_attn,
    float* __restrict__ outF)
{
  const int p = blockIdx.x;          // b*200 + q
  const int b = p / LL, q = p % LL;
  const int t = threadIdx.x;         // 0..63
  __shared__ float sM1[64], sb[64], scr[LL];
  sM1[t] = M1[(size_t)p * 64 + t];
  sb[t]  = b_attn[t];
  __syncthreads();
  for (int k = t; k <= q; k += 64) {
    const float* m2 = M2 + (size_t)(b*LL + k) * 64;
    float s = 0.f;
    #pragma unroll 8
    for (int d = 0; d < 64; ++d) s = fmaf(sb[d], sigm(sM1[d] + m2[d]), s);
    scr[k] = s;
  }
  __syncthreads();
  float acc = 0.f;
  for (int k = 0; k <= q; ++k) acc = fmaf(scr[k], seqs[(size_t)(b*LL + k) * 64 + t], acc);
  outF[(size_t)p * 64 + t] = acc;
}

// ---------------------------------------------------------------------------
// K6: PointWiseFeedForward + UpDown (one block of 128 per position)
// ---------------------------------------------------------------------------
__global__ __launch_bounds__(128) void k6_ffn(
    const float* __restrict__ F,
    const float* __restrict__ c1w, const float* __restrict__ c1b,
    const float* __restrict__ c2w, const float* __restrict__ c2b,
    const float* __restrict__ upw, const float* __restrict__ upb,
    const float* __restrict__ gw,  const float* __restrict__ gb,
    const float* __restrict__ dw,  const float* __restrict__ db,
    float* __restrict__ out)
{
  const int p = blockIdx.x, t = threadIdx.x;
  __shared__ float f[64], t1[64], f2[64], gu[128];
  if (t < 64) f[t] = F[(size_t)p * 64 + t];
  __syncthreads();
  if (t < 64) {
    float a = c1b[t];
    #pragma unroll
    for (int k = 0; k < 64; ++k) a = fmaf(f[k], c1w[k*64 + t], a);
    t1[t] = a > 0.f ? a : 0.f;
  }
  __syncthreads();
  if (t < 64) {
    float a = c2b[t];
    #pragma unroll
    for (int k = 0; k < 64; ++k) a = fmaf(t1[k], c2w[k*64 + t], a);
    f2[t] = f[t] + a;
  }
  __syncthreads();
  {
    float u = upb[t], g = gb[t];
    #pragma unroll
    for (int k = 0; k < 64; ++k) { const float x = f2[k]; u = fmaf(x, upw[k*128 + t], u); g = fmaf(x, gw[k*128 + t], g); }
    g = g > 0.f ? g : 0.f;
    gu[t] = g * u;
  }
  __syncthreads();
  if (t < 64) {
    float a = db[t];
    #pragma unroll
    for (int k = 0; k < 128; ++k) a = fmaf(gu[k], dw[k*64 + t], a);
    out[(size_t)p * 64 + t] = f2[t] + (a > 0.f ? a : 0.f);
  }
}

// ---------------------------------------------------------------------------
// K7: pos/neg logits
// ---------------------------------------------------------------------------
__global__ __launch_bounds__(256) void k7_logits(
    const float* __restrict__ F, const float* __restrict__ item_emb,
    const int* __restrict__ posq, const int* __restrict__ negq,
    float* __restrict__ out)
{
  __shared__ float rp[256], rn[256];
  const int t = threadIdx.x;
  const int p = blockIdx.x * 4 + (t >> 6);
  const int d = t & 63;
  const float fv = F[(size_t)p * 64 + d];
  rp[t] = fv * item_emb[(size_t)posq[p] * 64 + d];
  rn[t] = fv * item_emb[(size_t)negq[p] * 64 + d];
  __syncthreads();
  for (int s = 32; s > 0; s >>= 1) {
    if (d < s) { rp[t] += rp[t + s]; rn[t] += rn[t + s]; }
    __syncthreads();
  }
  if (d == 0) { out[p] = rp[t]; out[NPOS + p] = rn[t]; }
}

// ---------------------------------------------------------------------------
extern "C" void kernel_launch(void* const* d_in, const int* in_sizes, int n_in,
                              void* d_out, int out_size, void* d_ws, size_t ws_size,
                              hipStream_t stream) {
  (void)in_sizes; (void)n_in; (void)out_size; (void)ws_size;
  const int*   log_seqs = (const int*)  d_in[0];
  const int*   pos_seqs = (const int*)  d_in[1];
  const int*   neg_seqs = (const int*)  d_in[2];
  const float* adj      = (const float*)d_in[4];
  const float* item_emb = (const float*)d_in[5];
  const float* pos_emb  = (const float*)d_in[6];
  const float* W_item   = (const float*)d_in[7];
  const float* a_item   = (const float*)d_in[8];
  const float* W_1      = (const float*)d_in[9];
  const float* W_2      = (const float*)d_in[10];
  const float* b_attn   = (const float*)d_in[11];
  const float* co_c     = (const float*)d_in[12];
  const float* co_n     = (const float*)d_in[13];
  const float* c1w      = (const float*)d_in[14];
  const float* c1b      = (const float*)d_in[15];
  const float* c2w      = (const float*)d_in[16];
  const float* c2b      = (const float*)d_in[17];
  const float* upw      = (const float*)d_in[18];
  const float* upb      = (const float*)d_in[19];
  const float* gw       = (const float*)d_in[20];
  const float* gb       = (const float*)d_in[21];
  const float* dw       = (const float*)d_in[22];
  const float* db       = (const float*)d_in[23];

  char* w = (char*)d_ws;
  size_t off = 0;
  auto take = [&](size_t n) { size_t o = (off + 255) & ~(size_t)255; off = o + n; return o; };
  unsigned short* hT = (unsigned short*)(w + take((size_t)DDIM * NPAD * 2));
  float* wh1   = (float*)(w + take((size_t)NPAD * 4));
  float* wh2   = (float*)(w + take((size_t)NPAD * 4));
  float* trans = (float*)(w + take((size_t)NPAD * DDIM * 4));
  float* gat   = (float*)(w + take((size_t)NPAD * DDIM * 4));
  float* seqs  = (float*)(w + take((size_t)NPOS * DDIM * 4));
  float* seqsp = (float*)(w + take((size_t)NPOS * DDIM * 4));
  float* M1    = (float*)(w + take((size_t)NPOS * DDIM * 4));
  float* M2    = (float*)(w + take((size_t)NPOS * DDIM * 4));
  float* F1    = (float*)(w + take((size_t)NPOS * DDIM * 4));
  float* F2    = (float*)(w + take((size_t)NPOS * DDIM * 4));
  float* out   = (float*)d_out;

  k1_hproj <<<NPAD / 4, 256, 0, stream>>>(item_emb, W_item, a_item, hT, wh1, wh2);
  k2_gat   <<<NPAD / 16, 256, 0, stream>>>(adj, hT, wh1, wh2, trans, gat);
  k3_seqs  <<<NPOS / 4, 256, 0, stream>>>(log_seqs, item_emb, pos_emb, co_c, co_n, gat, trans, seqs, seqsp);
  k4_m12   <<<NPOS / 4, 256, 0, stream>>>(seqsp, W_1, W_2, M1, M2);
  k5_attn  <<<NPOS, 64, 0, stream>>>(M1, M2, seqs, b_attn, F1);
  k6_ffn   <<<NPOS, 128, 0, stream>>>(F1, c1w, c1b, c2w, c2b, upw, upb, gw, gb, dw, db, F2);
  k7_logits<<<NPOS / 4, 256, 0, stream>>>(F2, item_emb, pos_seqs, neg_seqs, out);
}
